// IGRNN_20899310862488
// MI455X (gfx1250) — matrix operand order repaired and checked by reading, not compiled, not run on hardware
//
#include <hip/hip_runtime.h>

// MI455X / gfx1250 persistent-GRU kernel.
// 32 workgroups x 128 threads (4 wave32). Each WG owns a 16-row batch tile and
// iterates all 256 timesteps with states in LDS. All GEMMs use
// v_wmma_f32_16x16x32_f16 (f16 in, f32 accumulate). Weights are converted once
// to padded f16 in d_ws and streamed from L2 as B fragments.

typedef _Float16 v16h __attribute__((ext_vector_type(16)));
typedef _Float16 v8h  __attribute__((ext_vector_type(8)));
typedef float    v8f  __attribute__((ext_vector_type(8)));

#define SEQLEN 256
#define BATCH  512
#define INDIM  138   // 32 robot + 49 p1 + 49 p2 + 8 goal

// Offsets (in _Float16 elements) of padded f16 weight blobs inside d_ws.
// All matrices stored row-major [N=3H or 16][K padded to mult of 32].
#define OFF_P1WX 0       // 192 x 64  (K 49 -> 64)
#define OFF_P1WH 12288   // 192 x 64
#define OFF_P2WX 24576   // 192 x 64
#define OFF_P2WH 36864   // 192 x 64
#define OFF_RCWX 49152   // 192 x 96
#define OFF_RCWH 67584   // 192 x 64
#define OFF_GCWX 79872   // 192 x 96  (K 80 -> 96)
#define OFF_GCWH 98304   // 192 x 64
#define OFF_FCW  110592  // 16  x 160 (K 136 -> 160)
#define W_TOTAL  113152  // halves (~226 KB)

__global__ void pad_convert(const float* __restrict__ src, _Float16* __restrict__ dst,
                            int rows, int scols, int dcols) {
    int i = blockIdx.x * blockDim.x + threadIdx.x;
    if (i >= rows * dcols) return;
    int r = i / dcols, c = i - r * dcols;
    dst[i] = (c < scols) ? (_Float16)src[r * scols + c] : (_Float16)0.0f;
}

// ---- WMMA fragment loaders (ISA 7.12.2 layouts, wave32) -------------------

// A: 16x32 f16 tile, row-major in LDS with stride lda. Lane l(0-15)=row M=l,
// hi half of wave takes K groups {8..15, 24..31}.
__device__ __forceinline__ v16h loadA(const _Float16* p, int lda, int lane) {
    int l = lane & 15, hi = (lane >> 4) & 1;
    const _Float16* r = p + l * lda + 8 * hi;
    v8h lo = *(const v8h*)r;          // K = 8*hi .. 8*hi+7
    v8h hh = *(const v8h*)(r + 16);   // K = 16+8*hi .. 16+8*hi+7
    v16h a;
#pragma unroll
    for (int i = 0; i < 8; ++i) { a[i] = lo[i]; a[8 + i] = hh[i]; }
    return a;
}

// B: 32x16 f16 tile where B[k][n] = W[n0+n][k0+k], W row-major [N][ldk] in
// global (L2-resident). Lane n(0-15)=col, upper lane half takes K groups {8..15, 24..31}.
// Addresses are 16B aligned -> two v8h loads.
__device__ __forceinline__ v16h loadB(const _Float16* __restrict__ W, int ldk,
                                      int n0, int k0, int lane) {
    int n = lane & 15, khi = (lane >> 4) & 1;
    const _Float16* r = W + (size_t)(n0 + n) * ldk + k0 + khi * 8;
    v8h lo = *(const v8h*)r;
    v8h hh = *(const v8h*)(r + 16);
    v16h b;
#pragma unroll
    for (int i = 0; i < 8; ++i) { b[i] = lo[i]; b[8 + i] = hh[i]; }
    return b;
}

// One 16x16 output tile of G[16][ldg] = A[16][32*kch] * W^T + bias.
__device__ __forceinline__ void gemm_tile(const _Float16* A, int lda, int kch,
                                          const _Float16* __restrict__ W, int ldk, int n0,
                                          const float* __restrict__ bias,
                                          float* G, int ldg, int lane) {
    float bv = bias[n0 + (lane & 15)];
    v8f acc = {bv, bv, bv, bv, bv, bv, bv, bv};
    for (int k = 0; k < kch; ++k) {
        v16h a = loadA(A + k * 32, lda, lane);
        v16h b = loadB(W, ldk, n0, k * 32, lane);
        acc = __builtin_amdgcn_wmma_f32_16x16x32_f16(
            /*neg_a=*/false, a, /*neg_b=*/false, b,
            /*c_mod=*/(short)0, acc, /*reuse_a=*/false, /*reuse_b=*/false);
    }
    int n = lane & 15, mh = (lane >> 4) & 1;
#pragma unroll
    for (int r = 0; r < 8; ++r) G[(r + 8 * mh) * ldg + n0 + n] = acc[r];
}

// Per-wave slice: 3 of the 12 N-tiles of the 192-wide gate matmul.
__device__ __forceinline__ void gemm3(const _Float16* A, int lda, int kch,
                                      const _Float16* __restrict__ W, int ldk,
                                      const float* __restrict__ bias,
                                      float* G, int wave, int lane) {
#pragma unroll
    for (int i = 0; i < 3; ++i)
        gemm_tile(A, lda, kch, W, ldk, (wave * 3 + i) * 16, bias, G, 192, lane);
}

// GRU gate fusion: h' = (1-z)*n + z*h; also mirrors h' as f16 into a concat
// staging buffer and optionally streams it to global output.
__device__ __forceinline__ void gru_update(const float* GX, const float* GH,
                                           _Float16* Hf,
                                           _Float16* dup, int dupld, int dupc,
                                           float* gout, int tid) {
    for (int idx = tid; idx < 16 * 64; idx += 128) {
        int m = idx >> 6, j = idx & 63;
        const float* gx = GX + m * 192;
        const float* gh = GH + m * 192;
        float r = 1.0f / (1.0f + __expf(-(gx[j]       + gh[j])));
        float z = 1.0f / (1.0f + __expf(-(gx[64 + j]  + gh[64 + j])));
        float n = tanhf(gx[128 + j] + r * gh[128 + j]);
        float h = (1.0f - z) * n + z * (float)Hf[idx];
        Hf[idx] = (_Float16)h;
        dup[m * dupld + dupc + j] = (_Float16)h;
        if (gout) gout[m * 64 + j] = h;
    }
}

struct __align__(16) Smem {
    _Float16 f_p1h[16 * 64];   // f16 hidden states (A fragments for h @ Wh^T)
    _Float16 f_p2h[16 * 64];
    _Float16 f_rh [16 * 64];
    _Float16 f_gh [16 * 64];
    _Float16 p1x  [16 * 64];   // x_p1 padded 49->64
    _Float16 p2x  [16 * 64];
    _Float16 rcin [16 * 96];   // [x_r | gh_prev]
    _Float16 gcin [16 * 96];   // [xt_g | rh | pad]
    _Float16 fcin [16 * 160];  // [p1h | p2h | x_g | pad]
    float    gA[16 * 192];     // input-side gates
    float    gB[16 * 192];     // hidden-side gates
    float    fcout[16 * 16];
};
#define ZERO_HALVES 11776      // all _Float16 members above (contiguous)

__global__ __launch_bounds__(128, 1)
void igrnn_persistent(const float* __restrict__ seq,
                      const _Float16* __restrict__ W,
                      const float* __restrict__ b_p1x, const float* __restrict__ b_p1h,
                      const float* __restrict__ b_p2x, const float* __restrict__ b_p2h,
                      const float* __restrict__ b_fc,
                      const float* __restrict__ b_rcx, const float* __restrict__ b_rch,
                      const float* __restrict__ b_gcx, const float* __restrict__ b_gch,
                      float* __restrict__ out) {
    __shared__ Smem sm;
    const int tid  = threadIdx.x;
    const int lane = tid & 31;
    const int wave = tid >> 5;
    const int b0   = blockIdx.x * 16;

    {   // zero hidden states + staging (pad columns stay zero forever)
        _Float16* z = sm.f_p1h;
        for (int i = tid; i < ZERO_HALVES; i += 128) z[i] = (_Float16)0.0f;
    }
    __syncthreads();

    for (int t = 0; t < SEQLEN; ++t) {
        // ---- scatter x_t slices into f16 staging ----
        const float* xt = seq + ((size_t)t * BATCH + b0) * INDIM;
        for (int i = tid; i < 16 * INDIM; i += 128) {
            int m = i / INDIM, c = i - m * INDIM;
            float v = xt[(size_t)m * INDIM + c];
            if      (c <  32) sm.rcin[m * 96 + c]              = (_Float16)v;
            else if (c <  81) sm.p1x [m * 64 + (c - 32)]       = (_Float16)v;
            else if (c < 130) sm.p2x [m * 64 + (c - 81)]       = (_Float16)v;
            else              sm.fcin[m * 160 + 128 + (c - 130)] = (_Float16)v;
        }
        if (t + 1 < SEQLEN) {   // prefetch next timestep tile into caches
            const float* xn = xt + (size_t)BATCH * INDIM;
            for (int i = tid * 32; i < 16 * INDIM; i += 128 * 32)
                __builtin_prefetch(xn + i, 0, 1);
        }
        __syncthreads();

        // ---- p1 GRU ----
        gemm3(sm.p1x,   64, 2, W + OFF_P1WX, 64, b_p1x, sm.gA, wave, lane);
        gemm3(sm.f_p1h, 64, 2, W + OFF_P1WH, 64, b_p1h, sm.gB, wave, lane);
        __syncthreads();
        gru_update(sm.gA, sm.gB, sm.f_p1h, sm.fcin, 160, 0, nullptr, tid);
        __syncthreads();

        // ---- p2 GRU ----
        gemm3(sm.p2x,   64, 2, W + OFF_P2WX, 64, b_p2x, sm.gA, wave, lane);
        gemm3(sm.f_p2h, 64, 2, W + OFF_P2WH, 64, b_p2h, sm.gB, wave, lane);
        __syncthreads();
        gru_update(sm.gA, sm.gB, sm.f_p2h, sm.fcin, 160, 64, nullptr, tid);
        __syncthreads();

        // ---- fc (wave 0, single N-tile) + rc GEMMs in parallel ----
        if (wave == 0)
            gemm_tile(sm.fcin, 160, 5, W + OFF_FCW, 160, 0, b_fc, sm.fcout, 16, lane);
        gemm3(sm.rcin, 96, 3, W + OFF_RCWX, 96, b_rcx, sm.gA, wave, lane);
        gemm3(sm.f_rh, 64, 2, W + OFF_RCWH, 64, b_rch, sm.gB, wave, lane);
        __syncthreads();
        // relu(fc) -> gcin[:, :16]
        for (int i = tid; i < 256; i += 128) {
            int m = i >> 4, n = i & 15;
            float v = sm.fcout[m * 16 + n];
            sm.gcin[m * 96 + n] = (_Float16)(v > 0.0f ? v : 0.0f);
        }
        // rc GRU update -> f_rh and gcin[:, 16:80]
        gru_update(sm.gA, sm.gB, sm.f_rh, sm.gcin, 96, 16, nullptr, tid);
        __syncthreads();

        // ---- gc GRU ----
        gemm3(sm.gcin, 96, 3, W + OFF_GCWX, 96, b_gcx, sm.gA, wave, lane);
        gemm3(sm.f_gh, 64, 2, W + OFF_GCWH, 64, b_gch, sm.gB, wave, lane);
        __syncthreads();
        // update -> f_gh, rcin[:, 32:96] (for next step), and global output
        gru_update(sm.gA, sm.gB, sm.f_gh, sm.rcin, 96, 32,
                   out + ((size_t)t * BATCH + b0) * 64, tid);
        __syncthreads();
    }
}

extern "C" void kernel_launch(void* const* d_in, const int* in_sizes, int n_in,
                              void* d_out, int out_size, void* d_ws, size_t ws_size,
                              hipStream_t stream) {
    (void)in_sizes; (void)n_in; (void)out_size; (void)ws_size;
    const float* seq   = (const float*)d_in[0];
    const float* p1_wx = (const float*)d_in[1];
    const float* p1_wh = (const float*)d_in[2];
    const float* p1_bx = (const float*)d_in[3];
    const float* p1_bh = (const float*)d_in[4];
    const float* p2_wx = (const float*)d_in[5];
    const float* p2_wh = (const float*)d_in[6];
    const float* p2_bx = (const float*)d_in[7];
    const float* p2_bh = (const float*)d_in[8];
    const float* fc_w  = (const float*)d_in[9];
    const float* fc_b  = (const float*)d_in[10];
    const float* rc_wx = (const float*)d_in[11];
    const float* rc_wh = (const float*)d_in[12];
    const float* rc_bx = (const float*)d_in[13];
    const float* rc_bh = (const float*)d_in[14];
    const float* gc_wx = (const float*)d_in[15];
    const float* gc_wh = (const float*)d_in[16];
    const float* gc_bx = (const float*)d_in[17];
    const float* gc_bh = (const float*)d_in[18];

    _Float16* W = (_Float16*)d_ws;
    auto pad = [&](const float* s, int off, int rows, int sc, int dc) {
        int total = rows * dc;
        pad_convert<<<(total + 255) / 256, 256, 0, stream>>>(s, W + off, rows, sc, dc);
    };
    pad(p1_wx, OFF_P1WX, 192, 49, 64);
    pad(p1_wh, OFF_P1WH, 192, 64, 64);
    pad(p2_wx, OFF_P2WX, 192, 49, 64);
    pad(p2_wh, OFF_P2WH, 192, 64, 64);
    pad(rc_wx, OFF_RCWX, 192, 96, 96);
    pad(rc_wh, OFF_RCWH, 192, 64, 64);
    pad(gc_wx, OFF_GCWX, 192, 80, 96);
    pad(gc_wh, OFF_GCWH, 192, 64, 64);
    pad(fc_w,  OFF_FCW,   16, 136, 160);

    igrnn_persistent<<<32, 128, 0, stream>>>(
        seq, W, p1_bx, p1_bh, p2_bx, p2_bh, fc_b,
        rc_bx, rc_bh, gc_bx, gc_bh, (float*)d_out);
}
